// Model_14250701488846
// MI455X (gfx1250) — matrix operand order repaired and checked by reading, not compiled, not run on hardware
//
#include <hip/hip_runtime.h>
#include <hip/hip_bf16.h>

typedef __attribute__((ext_vector_type(16))) _Float16 v16h;
typedef __attribute__((ext_vector_type(8)))  float    v8f;

#define BATCH 32
#define NATOM 2048
#define NNBR  12
#define FIN   64
#define FOUT  25
#define HSTRIDE 32                         // padded channel stride for h
#define HP_ELEMS (BATCH * NATOM * HSTRIDE) // padded h buffer floats

// ---------------------------------------------------------------------------
// Kernel 0: zero the accumulator region of the workspace (stats + coef + pool)
// ---------------------------------------------------------------------------
__global__ void k_zero(float* __restrict__ p, int n) {
    int i = blockIdx.x * blockDim.x + threadIdx.x;
    if (i < n) p[i] = 0.0f;
}

// Pack one 16x32 f16 A-fragment k-chunk from two contiguous 8-float runs.
__device__ __forceinline__ v16h pack_a(const float4 a0, const float4 a1,
                                       const float4 b0, const float4 b1) {
    v16h t;
    t[0]  = (_Float16)a0.x; t[1]  = (_Float16)a0.y;
    t[2]  = (_Float16)a0.z; t[3]  = (_Float16)a0.w;
    t[4]  = (_Float16)a1.x; t[5]  = (_Float16)a1.y;
    t[6]  = (_Float16)a1.z; t[7]  = (_Float16)a1.w;
    t[8]  = (_Float16)b0.x; t[9]  = (_Float16)b0.y;
    t[10] = (_Float16)b0.z; t[11] = (_Float16)b0.w;
    t[12] = (_Float16)b1.x; t[13] = (_Float16)b1.y;
    t[14] = (_Float16)b1.z; t[15] = (_Float16)b1.w;
    return t;
}

// ---------------------------------------------------------------------------
// Kernel 1: per-atom-type linear (64 -> 25) + ReLU via v_wmma_f32_16x16x32_f16
// One wave per 16-row tile, tiles aligned to each group's start.
// grid: (tile_stride, 3 groups, BATCH), block: 32 (one wave)
// ---------------------------------------------------------------------------
__global__ void k_type_linear(const float* __restrict__ atom,
                              const float* __restrict__ type_w,
                              const float* __restrict__ type_b,
                              float* __restrict__ h) {
    const int lane = threadIdx.x;
    const int g    = blockIdx.y;
    const int b    = blockIdx.z;
    const int gs[4] = {0, 683, 1366, 2048};
    const int s = gs[g], e = gs[g + 1];
    const int ntiles = (e - s + 15) >> 4;

    const int m    = lane & 15;
    const int half = lane >> 4;

    // ---- B fragments: W[64][25]; 2 K-chunks x 2 N-tiles.
    // Column index clamped: junk in cols >= 25 is stored only to h's pad cols 25..31.
    const float* W = type_w + g * FIN * FOUT;
    const int n0 = m;                                    // cols 0..15
    const int n1 = (16 + m < FOUT) ? 16 + m : FOUT - 1;  // clamped
    v16h bf[2][2];
    #pragma unroll
    for (int kc = 0; kc < 2; ++kc) {
        v16h t0, t1;
        #pragma unroll
        for (int i = 0; i < 16; ++i) {
            const int k = kc * 32 + ((i < 8) ? 8 * half + i : 16 + 8 * half + (i - 8)); // B layout: table order
            t0[i] = (_Float16)W[k * FOUT + n0];
            t1[i] = (_Float16)W[k * FOUT + n1];
        }
        bf[kc][0] = t0;
        bf[kc][1] = t1;
    }
    const float bias0 = type_b[g * FOUT + n0];
    const float bias1 = type_b[g * FOUT + n1];

    for (int t = blockIdx.x; t < ntiles; t += gridDim.x) {
        const int row0 = t * 16 + s;
        const int r    = row0 + m;
        const int rc   = (r < e) ? r : (e - 1);          // clamp: no exec games
        const float* Ap = atom + ((size_t)b * NATOM + (size_t)rc) * FIN + 8 * half;

        // ---- A fragments: four contiguous 8-float runs -> 8x b128 loads
        const float4* p0 = (const float4*)(Ap);          // k = 8*half + 0..7
        const float4* p1 = (const float4*)(Ap + 16);     // k = 8*half + 16..23
        const float4* p2 = (const float4*)(Ap + 32);     // k = 8*half + 32..39
        const float4* p3 = (const float4*)(Ap + 48);     // k = 8*half + 48..55
        v16h af0 = pack_a(p0[0], p0[1], p1[0], p1[1]);
        v16h af1 = pack_a(p2[0], p2[1], p3[0], p3[1]);

        v8f c0 = {}; v8f c1 = {};
        c0 = __builtin_amdgcn_wmma_f32_16x16x32_f16(false, af0, false, bf[0][0], (short)0, c0, false, false);
        c0 = __builtin_amdgcn_wmma_f32_16x16x32_f16(false, af1, false, bf[1][0], (short)0, c0, false, false);
        c1 = __builtin_amdgcn_wmma_f32_16x16x32_f16(false, af0, false, bf[0][1], (short)0, c1, false, false);
        c1 = __builtin_amdgcn_wmma_f32_16x16x32_f16(false, af1, false, bf[1][1], (short)0, c1, false, false);

        // ---- bias + ReLU + store; h row stride 32, cols 25..31 are junk pad
        float* Hb = h + (((size_t)b * NATOM + (size_t)row0 + 8 * half) << 5);
        if (row0 + 16 <= e) {                            // full tile: branch-free
            #pragma unroll
            for (int vr = 0; vr < 8; ++vr) {
                float* Hp = Hb + ((size_t)vr << 5);
                const float y0 = c0[vr] + bias0;
                const float y1 = c1[vr] + bias1;
                Hp[m]      = y0 > 0.0f ? y0 : 0.0f;
                Hp[16 + m] = y1 > 0.0f ? y1 : 0.0f;      // cols >=25: junk pad
            }
        } else {                                         // one tail tile per group
            #pragma unroll
            for (int vr = 0; vr < 8; ++vr) {
                const int rr = row0 + vr + 8 * half;
                if (rr < e) {
                    float* Hp = Hb + ((size_t)vr << 5);
                    const float y0 = c0[vr] + bias0;
                    const float y1 = c1[vr] + bias1;
                    Hp[m]      = y0 > 0.0f ? y0 : 0.0f;
                    Hp[16 + m] = y1 > 0.0f ? y1 : 0.0f;
                }
            }
        }
    }
}

// ---------------------------------------------------------------------------
// Kernel 2: BatchNorm statistics (sum, sumsq per channel) over B*N rows
// ---------------------------------------------------------------------------
__global__ void k_bn_stats(const float* __restrict__ h, float* __restrict__ stats) {
    __shared__ float ssum[32], ssq[32];
    if (threadIdx.x < 32) { ssum[threadIdx.x] = 0.0f; ssq[threadIdx.x] = 0.0f; }
    __syncthreads();

    float s[FOUT], q[FOUT];
    #pragma unroll
    for (int c = 0; c < FOUT; ++c) { s[c] = 0.0f; q[c] = 0.0f; }

    const int total = BATCH * NATOM;
    for (int r = blockIdx.x * blockDim.x + threadIdx.x; r < total;
         r += gridDim.x * blockDim.x) {
        const float4* Hp = (const float4*)(h + ((size_t)r << 5));
        #pragma unroll
        for (int v = 0; v < 6; ++v) {                    // channels 0..23
            const float4 x = Hp[v];
            s[4*v+0] += x.x; q[4*v+0] += x.x * x.x;
            s[4*v+1] += x.y; q[4*v+1] += x.y * x.y;
            s[4*v+2] += x.z; q[4*v+2] += x.z * x.z;
            s[4*v+3] += x.w; q[4*v+3] += x.w * x.w;
        }
        const float x24 = ((const float*)Hp)[24];
        s[24] += x24; q[24] += x24 * x24;
    }
    #pragma unroll
    for (int c = 0; c < FOUT; ++c) {
        atomicAdd(&ssum[c], s[c]);
        atomicAdd(&ssq[c],  q[c]);
    }
    __syncthreads();
    if (threadIdx.x < FOUT) {
        atomicAdd(&stats[threadIdx.x],      ssum[threadIdx.x]);
        atomicAdd(&stats[32 + threadIdx.x], ssq[threadIdx.x]);
    }
}

// ---------------------------------------------------------------------------
// Kernel 2b: fold stats + gamma/beta into per-channel scale/shift
// ---------------------------------------------------------------------------
__global__ void k_bn_coef(const float* __restrict__ stats,
                          const float* __restrict__ gamma,
                          const float* __restrict__ beta,
                          float* __restrict__ coef) {
    const int c = threadIdx.x;
    if (c < FOUT) {
        const float cnt  = (float)(BATCH * NATOM);
        const float mean = stats[c] / cnt;
        const float var  = stats[32 + c] / cnt - mean * mean;
        const float sc   = gamma[c] * rsqrtf(var + 1e-5f);
        coef[c]      = sc;
        coef[32 + c] = beta[c] - mean * sc;
    }
}

// ---------------------------------------------------------------------------
// Kernel 3: gather neighbors (BN fused), bond-weighted reduce, 25->25 WMMA,
//           ReLU, partial mean-pool via atomics. One wave per 16-atom tile.
// Each half-wave owns a 16-channel block of its row -> neighbor rows are
// fetched as 4x b128 loads (48 wide loads/lane vs ~150 scalar before).
// grid: (NATOM/16, BATCH), block: 32
// ---------------------------------------------------------------------------
__global__ void k_conv(const float* __restrict__ h,
                       const float* __restrict__ bond,
                       const int*   __restrict__ adj,
                       const float* __restrict__ coef,
                       const float* __restrict__ conv_w,
                       const float* __restrict__ conv_b,
                       float* __restrict__ pool) {
    __shared__ float msg[16][36];   // row-padded to 36 floats (bank skew)
    const int lane = threadIdx.x;
    const int b    = blockIdx.y;
    const int row0 = blockIdx.x * 16;
    const int m    = lane & 15;
    const int half = lane >> 4;
    const int rr   = row0 + m;      // always < NATOM (2048 = 128*16)
    const int cb   = half * 16;     // this lane's channel block: cb..cb+15

    // adj/bond rows are 48 B and 16-B aligned -> 3x b128 each
    const int4*   Ap = (const int4*)  (adj  + ((size_t)b * NATOM + (size_t)rr) * NNBR);
    const float4* Bp = (const float4*)(bond + ((size_t)b * NATOM + (size_t)rr) * NNBR);
    int   idx[NNBR];
    float w[NNBR];
    #pragma unroll
    for (int v = 0; v < 3; ++v) {
        const int4   iv = Ap[v];
        const float4 fv = Bp[v];
        idx[4*v+0] = iv.x; idx[4*v+1] = iv.y; idx[4*v+2] = iv.z; idx[4*v+3] = iv.w;
        w[4*v+0]   = fv.x; w[4*v+1]   = fv.y; w[4*v+2]   = fv.z; w[4*v+3]   = fv.w;
    }
    const float* Hbase = h + ((size_t)b * NATOM << 5);

    // BN coefs for this 16-channel block (pad channels 25..31 are zeroed)
    float4 sc4[4], sh4[4];
    #pragma unroll
    for (int v = 0; v < 4; ++v) {
        sc4[v] = *(const float4*)(coef + cb + 4 * v);
        sh4[v] = *(const float4*)(coef + 32 + cb + 4 * v);
    }

    // bond-weighted gather-reduce, BN fused; 4x b128 per neighbor row
    float4 acc[4];
    #pragma unroll
    for (int v = 0; v < 4; ++v) acc[v] = make_float4(0.f, 0.f, 0.f, 0.f);
    #pragma unroll
    for (int j = 0; j < NNBR; ++j) {
        const float4* Hv = (const float4*)(Hbase + ((size_t)idx[j] << 5) + cb);
        const float wj = w[j];
        #pragma unroll
        for (int v = 0; v < 4; ++v) {
            const float4 hv = Hv[v];
            acc[v].x += wj * (sc4[v].x * hv.x + sh4[v].x);
            acc[v].y += wj * (sc4[v].y * hv.y + sh4[v].y);
            acc[v].z += wj * (sc4[v].z * hv.z + sh4[v].z);
            acc[v].w += wj * (sc4[v].w * hv.w + sh4[v].w);
        }
    }
    #pragma unroll
    for (int v = 0; v < 4; ++v)
        *(float4*)&msg[m][cb + 4 * v] = acc[v];
    __syncthreads();

    // A fragment: two contiguous 8-float runs per lane (4x ds b128 reads).
    // Elements 0..7:  k = 8*half + 0..7   (always < 25)
    // Elements 8..15: k = 16 + 8*half + i (half=1 -> k=24..31, zero k>=25)
    const float* Mr = &msg[m][0];
    const float4 a0 = *(const float4*)(Mr + 8 * half);
    const float4 a1 = *(const float4*)(Mr + 8 * half + 4);
    const float4 b0 = *(const float4*)(Mr + 16 + 8 * half);
    const float4 b1 = *(const float4*)(Mr + 16 + 8 * half + 4);
    const bool lo = (half == 0);
    v16h af;
    af[0]  = (_Float16)a0.x; af[1]  = (_Float16)a0.y;
    af[2]  = (_Float16)a0.z; af[3]  = (_Float16)a0.w;
    af[4]  = (_Float16)a1.x; af[5]  = (_Float16)a1.y;
    af[6]  = (_Float16)a1.z; af[7]  = (_Float16)a1.w;
    af[8]  = (_Float16)b0.x;                       // k=16 or 24: valid
    af[9]  = (_Float16)(lo ? b0.y : 0.0f);
    af[10] = (_Float16)(lo ? b0.z : 0.0f);
    af[11] = (_Float16)(lo ? b0.w : 0.0f);
    af[12] = (_Float16)(lo ? b1.x : 0.0f);
    af[13] = (_Float16)(lo ? b1.y : 0.0f);
    af[14] = (_Float16)(lo ? b1.z : 0.0f);
    af[15] = (_Float16)(lo ? b1.w : 0.0f);

    // B fragments: conv_w [25][25] -> K zero-padded to 32, N clamped (junk ok)
    const int n1 = (16 + m < FOUT) ? 16 + m : FOUT - 1;
    v16h bf0, bf1;
    #pragma unroll
    for (int i = 0; i < 16; ++i) {
        const int k  = (i < 8) ? 8 * half + i : 16 + 8 * half + (i - 8);
        const int kc = (k < FOUT) ? k : FOUT - 1;        // clamped load
        const float w0 = conv_w[kc * FOUT + m];
        const float w1 = conv_w[kc * FOUT + n1];
        bf0[i] = (_Float16)((k < FOUT) ? w0 : 0.0f);     // select, not branch
        bf1[i] = (_Float16)((k < FOUT) ? w1 : 0.0f);
    }

    v8f d0 = {}; v8f d1 = {};
    d0 = __builtin_amdgcn_wmma_f32_16x16x32_f16(false, af, false, bf0, (short)0, d0, false, false);
    d1 = __builtin_amdgcn_wmma_f32_16x16x32_f16(false, af, false, bf1, (short)0, d1, false, false);

    // bias + ReLU, sum the 8 rows this lane holds, accumulate pool[b][c]
    const float cb0 = conv_b[m];
    const float cb1 = conv_b[n1];
    float p0 = 0.0f, p1 = 0.0f;
    #pragma unroll
    for (int vr = 0; vr < 8; ++vr) {
        const float y0 = d0[vr] + cb0;
        p0 += y0 > 0.0f ? y0 : 0.0f;
        const float y1 = d1[vr] + cb1;
        p1 += y1 > 0.0f ? y1 : 0.0f;
    }
    atomicAdd(&pool[b * FOUT + m], p0);
    if (16 + m < FOUT) atomicAdd(&pool[b * FOUT + 16 + m], p1);
}

// ---------------------------------------------------------------------------
// Kernel 4: e[b] = relu( (pool[b]/N) . out_w + out_b )
// ---------------------------------------------------------------------------
__global__ void k_final(const float* __restrict__ pool,
                        const float* __restrict__ out_w,
                        const float* __restrict__ out_b,
                        float* __restrict__ out) {
    const int b = threadIdx.x;
    if (b < BATCH) {
        float s = 0.0f;
        #pragma unroll
        for (int c = 0; c < FOUT; ++c) s += pool[b * FOUT + c] * out_w[c];
        s = s * (1.0f / (float)NATOM) + out_b[0];
        out[b] = s > 0.0f ? s : 0.0f;
    }
}

// ---------------------------------------------------------------------------
extern "C" void kernel_launch(void* const* d_in, const int* in_sizes, int n_in,
                              void* d_out, int out_size, void* d_ws, size_t ws_size,
                              hipStream_t stream) {
    const float* atom   = (const float*)d_in[0];
    const float* bond   = (const float*)d_in[1];
    const int*   adj    = (const int*)  d_in[2];
    const float* type_w = (const float*)d_in[3];
    const float* type_b = (const float*)d_in[4];
    const float* gamma  = (const float*)d_in[5];
    const float* beta   = (const float*)d_in[6];
    const float* conv_w = (const float*)d_in[7];
    const float* conv_b = (const float*)d_in[8];
    const float* out_w  = (const float*)d_in[9];
    const float* out_b  = (const float*)d_in[10];
    float* out = (float*)d_out;

    float* ws    = (float*)d_ws;
    float* h     = ws;                    // [B, N, 32] (25 valid + 7 pad)
    float* stats = ws + HP_ELEMS;         // sum[0..24], sumsq[32..56]
    float* coef  = stats + 64;            // scale[0..24 | 25..31=0], shift[32..56 | 57..63=0]
    float* pool  = coef + 64;             // [B, 25]

    // zero stats + coef + pool each call — graph-replay safe
    k_zero<<<4, 256, 0, stream>>>(stats, 64 + 64 + BATCH * FOUT);

    k_type_linear<<<dim3(8, 3, BATCH), 32, 0, stream>>>(atom, type_w, type_b, h);
    k_bn_stats<<<128, 256, 0, stream>>>(h, stats);
    k_bn_coef<<<1, 32, 0, stream>>>(stats, gamma, beta, coef);
    k_conv<<<dim3(NATOM / 16, BATCH), 32, 0, stream>>>(h, bond, adj, coef,
                                                       conv_w, conv_b, pool);
    k_final<<<1, 32, 0, stream>>>(pool, out_w, out_b, out);
}
